// LSTMDecoder_249108103774
// MI455X (gfx1250) — compile-verified, matrix operand order checked by reading, not run on hardware
//
#include <hip/hip_runtime.h>
#include <hip/hip_bf16.h>
#include <math.h>

typedef __attribute__((ext_vector_type(16))) _Float16 v16h;
typedef __attribute__((ext_vector_type(8)))  float    v8f;

#define BATCH 16384
#define EMB   64
#define HID   512
#define INP   11
#define NSTEP 5
#define KC    (EMB + HID)   // 576

__device__ __forceinline__ float sigmf(float x) { return 1.0f / (1.0f + __expf(-x)); }

// Generic pointer -> 32-bit LDS byte offset (LDS aperture: offset is the low 32 address bits)
__device__ __forceinline__ unsigned lds_off_of(const void* p) {
  return (unsigned)(unsigned long long)p;
}

// Async 16B copy global -> LDS (per-lane), tracked by ASYNCcnt
__device__ __forceinline__ void async_cp16(unsigned lds_off, const _Float16* g) {
  asm volatile("global_load_async_to_lds_b128 %0, %1, off"
               :: "v"(lds_off), "v"(g) : "memory");
}

__device__ __forceinline__ void wait_async0() {
  asm volatile("s_wait_asynccnt 0x0" ::: "memory");
}

// ---------------------------------------------------------------------------
// Init: zero states, convert embedding to f16
// ---------------------------------------------------------------------------
__global__ void lstm_init_kernel(const float* __restrict__ emb,
                                 _Float16* __restrict__ x16,
                                 _Float16* __restrict__ h0,
                                 float* __restrict__ c0,
                                 float* __restrict__ h1,
                                 float* __restrict__ c1) {
  int i = blockIdx.x * blockDim.x + threadIdx.x;
  if (i < BATCH * HID) { c0[i] = 0.0f; h0[i] = (_Float16)0.0f; }
  if (i < BATCH * EMB) { x16[i] = (_Float16)emb[i]; }
  if (i < BATCH * INP) { h1[i] = 0.0f; c1[i] = 0.0f; }
}

// ---------------------------------------------------------------------------
// Weight prep: Wcat1[g][0:64]=Wih1, [64:576]=Whh1 (f16); b1 = bih1 + bhh1
// ---------------------------------------------------------------------------
__global__ void lstm_wprep_kernel(const float* __restrict__ Wih1,
                                  const float* __restrict__ Whh1,
                                  const float* __restrict__ bih1,
                                  const float* __restrict__ bhh1,
                                  _Float16* __restrict__ Wcat,
                                  float* __restrict__ b1) {
  int i = blockIdx.x * blockDim.x + threadIdx.x;
  if (i < 4 * HID * KC) {
    int g = i / KC, k = i % KC;
    float v = (k < EMB) ? Wih1[g * EMB + k] : Whh1[g * HID + (k - EMB)];
    Wcat[i] = (_Float16)v;
  }
  if (i < 4 * HID) b1[i] = bih1[i] + bhh1[i];
}

// ---------------------------------------------------------------------------
// Layer-1 LSTM cell: fused GEMM(WMMA f16) + gates + cell update.
// Block tile: 128 rows x 32 hid cols; 8 waves, each wave = 16-row strip with
// 4 gates x 2 n-tiles accumulators. Async global->LDS staging, double buffer.
// Grid: 2048 blocks x 256 threads.
// ---------------------------------------------------------------------------
__device__ __forceinline__ void stage_chunk_async(int kc, int tid, int m0, int n0,
                                                  const _Float16* __restrict__ x16,
                                                  const _Float16* __restrict__ h0r,
                                                  const _Float16* __restrict__ Wcat,
                                                  _Float16* As, _Float16* Bs) {
  const int k0 = kc * 32;
  // A: 128 rows x 32 halves = 512 x 16B segments, 2 per thread
#pragma unroll
  for (int j = 0; j < 2; ++j) {
    int s   = tid + j * 256;
    int row = s >> 2;
    int off = (s & 3) * 8;
    const _Float16* src = (kc < 2)
        ? x16 + (size_t)(m0 + row) * EMB + k0 + off
        : h0r + (size_t)(m0 + row) * HID + (k0 - EMB) + off;
    async_cp16(lds_off_of(&As[row * 40 + off]), src);
  }
  // B: 4 gates x 32 cols x 32 halves = 512 x 16B segments, 2 per thread
#pragma unroll
  for (int j = 0; j < 2; ++j) {
    int s    = tid + j * 256;
    int rsel = s >> 2;
    int g    = rsel >> 5;
    int n    = rsel & 31;
    int off  = (s & 3) * 8;
    const _Float16* src = Wcat + (size_t)(g * HID + n0 + n) * KC + k0 + off;
    async_cp16(lds_off_of(&Bs[(g * 32 + n) * 40 + off]), src);
  }
}

__global__ __launch_bounds__(256)
void lstm_cell1_wmma_kernel(const _Float16* __restrict__ x16,
                            const _Float16* __restrict__ h0r,
                            _Float16* __restrict__ h0w,
                            float* __restrict__ c0,
                            const _Float16* __restrict__ Wcat,
                            const float* __restrict__ b1) {
  __shared__ __attribute__((aligned(16))) _Float16 As[2][128 * 40];     // [buf][row][k]
  __shared__ __attribute__((aligned(16))) _Float16 Bs[2][4 * 32 * 40];  // [buf][gate*32+n][k]

  const int tid  = threadIdx.x;
  const int wave = tid >> 5;
  const int lane = tid & 31;
  const int lrow = lane & 15;
  const int khi8 = (lane >> 4) << 3;  // 0 or 8: per-lane K sub-offset (ISA fragment layout)

  const int n_block = blockIdx.x & 15;   // 16 blocks of 32 hid cols
  const int m_block = blockIdx.x >> 4;   // 128 blocks of 128 rows
  const int m0 = m_block * 128;
  const int n0 = n_block * 32;

  // Bias folded into accumulator init (same bias for all M in a column)
  v8f acc[4][2];
#pragma unroll
  for (int g = 0; g < 4; ++g)
#pragma unroll
    for (int nt = 0; nt < 2; ++nt) {
      float bv = b1[g * HID + n0 + nt * 16 + lrow];
#pragma unroll
      for (int e = 0; e < 8; ++e) acc[g][nt][e] = bv;
    }

  // Prologue: stage chunk 0
  stage_chunk_async(0, tid, m0, n0, x16, h0r, Wcat, As[0], Bs[0]);
  wait_async0();
  __syncthreads();

  const int NKC = KC / 32;  // 18
  for (int kc = 0; kc < NKC; ++kc) {
    const int cur = kc & 1;
    const int nxt = cur ^ 1;
    // Issue next chunk's async copies into the other buffer (overlaps WMMA)
    if (kc + 1 < NKC)
      stage_chunk_async(kc + 1, tid, m0, n0, x16, h0r, Wcat, As[nxt], Bs[nxt]);

    // ---- A fragment (16x32 f16, per-lane packing per ISA table)
    v16h afrag;
    const int arow = wave * 16 + lrow;
#pragma unroll
    for (int v = 0; v < 8; ++v) {
      int k = ((v >> 2) << 4) + khi8 + ((v & 3) << 1);
      afrag[2 * v]     = As[cur][arow * 40 + k];
      afrag[2 * v + 1] = As[cur][arow * 40 + k + 1];
    }
    // ---- 4 gates x 2 n-tiles WMMAs
#pragma unroll
    for (int g = 0; g < 4; ++g)
#pragma unroll
      for (int nt = 0; nt < 2; ++nt) {
        v16h bfrag;
        const int bbase = (g * 32 + nt * 16 + lrow) * 40;
#pragma unroll
        for (int v = 0; v < 8; ++v) {
          int k = ((v >> 2) << 4) + khi8 + ((v & 3) << 1);
          bfrag[2 * v]     = Bs[cur][bbase + k];
          bfrag[2 * v + 1] = Bs[cur][bbase + k + 1];
        }
        acc[g][nt] = __builtin_amdgcn_wmma_f32_16x16x32_f16(
            false, afrag, false, bfrag, (short)0, acc[g][nt], false, false);
      }

    // Ensure next buffer fully landed, and all waves done before reuse
    wait_async0();
    __syncthreads();
  }

  // ---- fused LSTM cell epilogue (C/D layout: M = v + khi8, N = lane&15)
#pragma unroll
  for (int nt = 0; nt < 2; ++nt) {
    const int n = n0 + nt * 16 + lrow;
#pragma unroll
    for (int v = 0; v < 8; ++v) {
      int m = m0 + wave * 16 + v + khi8;
      size_t idx = (size_t)m * HID + n;
      float gi = sigmf(acc[0][nt][v]);
      float gf = sigmf(acc[1][nt][v]);
      float gg = tanhf(acc[2][nt][v]);
      float go = sigmf(acc[3][nt][v]);
      float cn = fmaf(gf, c0[idx], gi * gg);
      c0[idx] = cn;
      h0w[idx] = (_Float16)(go * tanhf(cn));
    }
  }
}

// ---------------------------------------------------------------------------
// Layer-2 LSTM cell (tiny: N=44, K=523) — scalar VALU, fused update + output
// ---------------------------------------------------------------------------
__global__ __launch_bounds__(256)
void lstm_cell2_kernel(const _Float16* __restrict__ h0,
                       const float* __restrict__ h1r,
                       float* __restrict__ h1w,
                       float* __restrict__ c1,
                       const float* __restrict__ Wih2,
                       const float* __restrict__ Whh2,
                       const float* __restrict__ bih2,
                       const float* __restrict__ bhh2,
                       float* __restrict__ out_step) {
  int t = blockIdx.x * blockDim.x + threadIdx.x;
  if (t >= BATCH * INP) return;
  int b = t / INP;
  int n = t % INP;

  float acc[4];
#pragma unroll
  for (int g = 0; g < 4; ++g) {
    int col = g * INP + n;
    acc[g] = bih2[col] + bhh2[col];
  }
  const _Float16* hrow = h0 + (size_t)b * HID;
  for (int k = 0; k < HID; ++k) {
    float hv = (float)hrow[k];
#pragma unroll
    for (int g = 0; g < 4; ++g)
      acc[g] = fmaf(hv, Wih2[(size_t)(g * INP + n) * HID + k], acc[g]);
  }
#pragma unroll
  for (int k = 0; k < INP; ++k) {
    float hv = h1r[b * INP + k];
#pragma unroll
    for (int g = 0; g < 4; ++g)
      acc[g] = fmaf(hv, Whh2[(g * INP + n) * INP + k], acc[g]);
  }
  float gi = sigmf(acc[0]);
  float gf = sigmf(acc[1]);
  float gg = tanhf(acc[2]);
  float go = sigmf(acc[3]);
  float cn = fmaf(gf, c1[t], gi * gg);
  c1[t] = cn;
  float h = go * tanhf(cn);
  h1w[t] = h;
  out_step[t] = h;
}

// ---------------------------------------------------------------------------
extern "C" void kernel_launch(void* const* d_in, const int* in_sizes, int n_in,
                              void* d_out, int out_size, void* d_ws, size_t ws_size,
                              hipStream_t stream) {
  const float* emb  = (const float*)d_in[0];
  const float* Wih1 = (const float*)d_in[1];
  const float* Whh1 = (const float*)d_in[2];
  const float* bih1 = (const float*)d_in[3];
  const float* bhh1 = (const float*)d_in[4];
  const float* Wih2 = (const float*)d_in[5];
  const float* Whh2 = (const float*)d_in[6];
  const float* bih2 = (const float*)d_in[7];
  const float* bhh2 = (const float*)d_in[8];
  float* out = (float*)d_out;
  (void)in_sizes; (void)n_in; (void)out_size; (void)ws_size;

  char* ws = (char*)d_ws;
  size_t off = 0;
  auto alloc = [&](size_t bytes) -> char* {
    char* p = ws + off;
    off += (bytes + 255) & ~(size_t)255;
    return p;
  };
  _Float16* x16  = (_Float16*)alloc((size_t)BATCH * EMB * 2);
  _Float16* h0A  = (_Float16*)alloc((size_t)BATCH * HID * 2);
  _Float16* h0B  = (_Float16*)alloc((size_t)BATCH * HID * 2);
  float*    c0   = (float*)   alloc((size_t)BATCH * HID * 4);
  float*    h1A  = (float*)   alloc((size_t)BATCH * INP * 4);
  float*    h1B  = (float*)   alloc((size_t)BATCH * INP * 4);
  float*    c1   = (float*)   alloc((size_t)BATCH * INP * 4);
  _Float16* Wcat = (_Float16*)alloc((size_t)4 * HID * KC * 2);
  float*    b1   = (float*)   alloc((size_t)4 * HID * 4);

  {
    int n = BATCH * HID;
    lstm_init_kernel<<<(n + 255) / 256, 256, 0, stream>>>(emb, x16, h0A, c0, h1A, c1);
  }
  {
    int n = 4 * HID * KC;
    lstm_wprep_kernel<<<(n + 255) / 256, 256, 0, stream>>>(Wih1, Whh1, bih1, bhh1, Wcat, b1);
  }
  for (int step = 0; step < NSTEP; ++step) {
    const _Float16* h0r = (step & 1) ? h0B : h0A;
    _Float16*       h0w = (step & 1) ? h0A : h0B;
    const float*    h1r = (step & 1) ? h1B : h1A;
    float*          h1w = (step & 1) ? h1A : h1B;
    lstm_cell1_wmma_kernel<<<dim3(2048), dim3(256), 0, stream>>>(x16, h0r, h0w, c0, Wcat, b1);
    lstm_cell2_kernel<<<dim3((BATCH * INP + 255) / 256), dim3(256), 0, stream>>>(
        h0w, h1r, h1w, c1, Wih2, Whh2, bih2, bhh2, out + (size_t)step * BATCH * INP);
  }
}
